// MultiHeadSelfAttention_54803782697214
// MI455X (gfx1250) — compile-verified, matrix operand order checked by reading, not run on hardware
//
#include <hip/hip_runtime.h>

// ---------------------------------------------------------------- constants
#define DIMC    768
#define NHEADS  12
#define HDIM    64
#define BATCH   2
#define SEQ     2048
#define MROWS   (BATCH*SEQ)        // 4096
#define K3      (3*DIMC)           // 2304

typedef __attribute__((ext_vector_type(16))) __bf16 v16bf;
typedef __attribute__((ext_vector_type(8)))  float  v8f;

union Frag16 {            // one 16x32 bf16 WMMA A/B fragment (8 VGPRs)
    v16bf v;
    uint4 u[2];
};

// async 16-byte global -> LDS copy (per active lane), ASYNCcnt-tracked.
// ldsoff: LDS byte address (low 32 bits of generic shared pointer)
// goff  : byte offset from 64-bit SGPR base (GVS addressing)
__device__ __forceinline__ void async_copy_b128(unsigned ldsoff, unsigned goff,
                                                unsigned long long base) {
    asm volatile("global_load_async_to_lds_b128 %0, %1, %2 offset:0"
                 :: "v"(ldsoff), "v"(goff), "s"(base) : "memory");
}
__device__ __forceinline__ void async_wait0() {
    asm volatile("s_wait_asynccnt 0" ::: "memory");
}

// workspace layout (bytes)
static constexpr size_t XB_OFF    = 0;                                   // x bf16   [4096,768]
static constexpr size_t XB_BYTES  = (size_t)MROWS * DIMC * 2;
static constexpr size_t WQKV_OFF  = XB_OFF + XB_BYTES;                   // qkv_w bf16 [2304,768]
static constexpr size_t WQKV_BYTES= (size_t)K3 * DIMC * 2;
static constexpr size_t WPRJ_OFF  = WQKV_OFF + WQKV_BYTES;               // proj_w bf16 [768,768]
static constexpr size_t WPRJ_BYTES= (size_t)DIMC * DIMC * 2;
static constexpr size_t QKV_OFF   = WPRJ_OFF + WPRJ_BYTES;               // q,k,vT bf16
static constexpr size_t QKV_ELEMS = (size_t)BATCH * NHEADS * SEQ * HDIM; // 3,145,728 per tensor
static constexpr size_t QKV_BYTES = 3 * QKV_ELEMS * 2;
static constexpr size_t OBUF_OFF  = QKV_OFF + QKV_BYTES;                 // attn out bf16 [4096,768]

// ---------------------------------------------------------------- fp32 -> bf16
__global__ __launch_bounds__(256) void f32_to_bf16_kernel(const float* __restrict__ in,
                                                          __bf16* __restrict__ out, int n) {
    int i = (blockIdx.x * 256 + threadIdx.x) * 4;
    if (i < n) {
        float4 f = *(const float4*)(in + i);
        union { __bf16 b[4]; uint2 u; } pk;
        pk.b[0] = (__bf16)f.x; pk.b[1] = (__bf16)f.y;
        pk.b[2] = (__bf16)f.z; pk.b[3] = (__bf16)f.w;
        *(uint2*)(out + i) = pk.u;
    }
}

// ---------------------------------------------------------------- WMMA GEMM
// C[M,Nn] = A[M,K] (bf16, row-major) @ Bw[Nn,K]^T (bf16, row-major) + bias
// EPI==0 : write fp32 to outF[M,Nn]
// EPI==1 : scatter QKV: q (scaled 0.125) -> [B,H,N,64], k -> [B,H,N,64], v -> [B,H,64,N]
template <int EPI>
__global__ __launch_bounds__(256) void gemm_bf16_wmma(const __bf16* __restrict__ A,
                                                      const __bf16* __restrict__ Bw,
                                                      const float* __restrict__ bias,
                                                      float* __restrict__ outF,
                                                      __bf16* __restrict__ outQKV,
                                                      int M, int Nn, int K) {
    constexpr int BK = 32;
    __shared__ __bf16 As[128 * BK];   // 8 KB
    __shared__ __bf16 Bs[128 * BK];   // 8 KB

    const int tid    = threadIdx.x;
    const int lane   = tid & 31;
    const int wid    = tid >> 5;          // 0..7
    const int lane15 = lane & 15;
    const int h8     = lane >> 4;         // 0/1 -> K-half select per ISA A layout
    const int waveM  = wid >> 2;          // 0..1  (64-row strip)
    const int waveN  = wid & 3;           // 0..3  (32-col strip)
    const int bm     = blockIdx.y * 128;
    const int bn     = blockIdx.x * 128;

    const unsigned long long aBase = (unsigned long long)(uintptr_t)A;
    const unsigned long long bBase = (unsigned long long)(uintptr_t)Bw;

    v8f acc[4][2];
#pragma unroll
    for (int mt = 0; mt < 4; ++mt)
#pragma unroll
        for (int nt = 0; nt < 2; ++nt)
#pragma unroll
            for (int e = 0; e < 8; ++e) acc[mt][nt][e] = 0.0f;

    for (int k0 = 0; k0 < K; k0 += BK) {
        // async global -> LDS tile stage (no VGPR round-trip), 16B per lane
#pragma unroll
        for (int t = 0; t < 2; ++t) {
            int c   = tid + t * 256;          // 0..511
            int row = c >> 2;
            int cc  = (c & 3) * 8;
            unsigned lA = (unsigned)(uintptr_t)&As[row * BK + cc];
            unsigned gA = (unsigned)(((unsigned)(bm + row) * K + k0 + cc) * 2u);
            async_copy_b128(lA, gA, aBase);
            unsigned lB = (unsigned)(uintptr_t)&Bs[row * BK + cc];
            unsigned gB = (unsigned)(((unsigned)(bn + row) * K + k0 + cc) * 2u);
            async_copy_b128(lB, gB, bBase);
        }
        if (k0 + BK < K) {  // warm L2 for next tile -> global_prefetch_b8
            __builtin_prefetch(&A [(size_t)(bm + (tid >> 1)) * K + k0 + BK], 0, 1);
            __builtin_prefetch(&Bw[(size_t)(bn + (tid >> 1)) * K + k0 + BK], 0, 1);
        }
        async_wait0();
        __syncthreads();

        Frag16 af[4], bf[2];
#pragma unroll
        for (int mt = 0; mt < 4; ++mt) {
            const __bf16* ar = &As[(waveM * 64 + mt * 16 + lane15) * BK];
            af[mt].u[0] = *(const uint4*)&ar[h8 * 8];
            af[mt].u[1] = *(const uint4*)&ar[16 + h8 * 8];
        }
#pragma unroll
        for (int nt = 0; nt < 2; ++nt) {
            const __bf16* br = &Bs[(waveN * 32 + nt * 16 + lane15) * BK];
            bf[nt].u[0] = *(const uint4*)&br[h8 * 8];
            bf[nt].u[1] = *(const uint4*)&br[16 + h8 * 8];
        }
#pragma unroll
        for (int mt = 0; mt < 4; ++mt)
#pragma unroll
            for (int nt = 0; nt < 2; ++nt)
                acc[mt][nt] = __builtin_amdgcn_wmma_f32_16x16x32_bf16(
                    false, af[mt].v, false, bf[nt].v, (short)0, acc[mt][nt], false, false);
        __syncthreads();
    }

    // ------------------------------------------------------------ epilogue
#pragma unroll
    for (int nt = 0; nt < 2; ++nt) {
        const int n  = bn + waveN * 32 + nt * 16 + lane15;
        const float bv = bias[n];
        if (EPI == 0) {
#pragma unroll
            for (int mt = 0; mt < 4; ++mt)
#pragma unroll
                for (int r = 0; r < 8; ++r) {
                    int m = bm + waveM * 64 + mt * 16 + r + 8 * h8;
                    outF[(size_t)m * Nn + n] = acc[mt][nt][r] + bv;
                }
        } else {
            const int which = n / DIMC;
            const int rem   = n - which * DIMC;
            const int hh    = rem >> 6;
            const int dd    = rem & 63;
#pragma unroll
            for (int mt = 0; mt < 4; ++mt)
#pragma unroll
                for (int r = 0; r < 8; ++r) {
                    int   m  = bm + waveM * 64 + mt * 16 + r + 8 * h8;
                    int   bb = m >> 11;
                    int   ns = m & (SEQ - 1);
                    float vv = acc[mt][nt][r] + bv;
                    if (which == 0) {        // q, pre-scaled by HDIM^-0.5
                        outQKV[((size_t)(bb * NHEADS + hh) * SEQ + ns) * HDIM + dd] =
                            (__bf16)(vv * 0.125f);
                    } else if (which == 1) { // k
                        outQKV[QKV_ELEMS + ((size_t)(bb * NHEADS + hh) * SEQ + ns) * HDIM + dd] =
                            (__bf16)vv;
                    } else {                 // v transposed -> [B,H,64,N]
                        outQKV[2 * QKV_ELEMS + ((size_t)(bb * NHEADS + hh) * HDIM + dd) * SEQ + ns] =
                            (__bf16)vv;
                    }
                }
        }
    }
}

// ---------------------------------------------------------------- flash attention
// grid = (SEQ/128, BATCH*NHEADS), block = 256 (8 waves); wave owns a 16-row Q strip
__global__ __launch_bounds__(256) void flash_attn_wmma(const __bf16* __restrict__ qbuf,
                                                       const __bf16* __restrict__ kbuf,
                                                       const __bf16* __restrict__ vt,
                                                       __bf16* __restrict__ obuf) {
    __shared__ __bf16 Ks[128 * 64];      // 16 KB : K rows for this block
    __shared__ __bf16 Vs[64 * 128];      // 16 KB : V^T rows (d major)
    __shared__ __bf16 Ps[8 * 16 * 128];  // 32 KB : per-wave P strip

    const int tid    = threadIdx.x;
    const int lane   = tid & 31;
    const int wid    = tid >> 5;
    const int lane15 = lane & 15;
    const int h8     = lane >> 4;
    const int bh     = blockIdx.y;               // b*NHEADS + h
    const int b      = bh / NHEADS;
    const int h      = bh - b * NHEADS;
    const int qblk   = blockIdx.x * 128;
    const int qr0    = qblk + wid * 16;          // this wave's first Q row

    const unsigned long long kBase = (unsigned long long)(uintptr_t)kbuf;
    const unsigned long long vBase = (unsigned long long)(uintptr_t)vt;

    // Q fragments (16 rows x 64) resident in registers
    Frag16 qf0, qf1;
    {
        const __bf16* qr = qbuf + ((size_t)bh * SEQ + qr0 + lane15) * HDIM;
        qf0.u[0] = *(const uint4*)&qr[h8 * 8];
        qf0.u[1] = *(const uint4*)&qr[16 + h8 * 8];
        qf1.u[0] = *(const uint4*)&qr[32 + h8 * 8];
        qf1.u[1] = *(const uint4*)&qr[48 + h8 * 8];
    }

    float m_i[8], l_i[8];
    v8f   o[4];
#pragma unroll
    for (int r = 0; r < 8; ++r) { m_i[r] = -1e30f; l_i[r] = 0.0f; }
#pragma unroll
    for (int dt = 0; dt < 4; ++dt)
#pragma unroll
        for (int e = 0; e < 8; ++e) o[dt][e] = 0.0f;

    for (int kb = 0; kb < SEQ / 128; ++kb) {
        const int kc0 = kb * 128;
        __syncthreads();   // previous iteration done with Ks/Vs
        // async global -> LDS stage of K tile [128,64] and V^T tile [64,128]
#pragma unroll
        for (int t = 0; t < 4; ++t) {
            int c  = tid + t * 256;               // 0..1023
            int kr = c >> 3, k8 = (c & 7) * 8;
            unsigned lK = (unsigned)(uintptr_t)&Ks[kr * 64 + k8];
            unsigned gK = (unsigned)((((unsigned)bh * SEQ + kc0 + kr) * HDIM + k8) * 2u);
            async_copy_b128(lK, gK, kBase);
            int vr = c >> 4, v8c = (c & 15) * 8;
            unsigned lV = (unsigned)(uintptr_t)&Vs[vr * 128 + v8c];
            unsigned gV = (unsigned)((((unsigned)bh * HDIM + vr) * SEQ + kc0 + v8c) * 2u);
            async_copy_b128(lV, gV, vBase);
        }
        if (kb + 1 < SEQ / 128) {
            __builtin_prefetch(&kbuf[((size_t)bh * SEQ + kc0 + 128 + (tid >> 1)) * HDIM], 0, 1);
            __builtin_prefetch(&vt[((size_t)bh * HDIM + (tid >> 2)) * SEQ + kc0 + 128], 0, 1);
        }
        async_wait0();
        __syncthreads();

        // S = Q @ K^T for 16x128 strip (Q pre-scaled by 1/sqrt(D))
        v8f sacc[8];
#pragma unroll
        for (int t = 0; t < 8; ++t) {
#pragma unroll
            for (int e = 0; e < 8; ++e) sacc[t][e] = 0.0f;
            const __bf16* krow = &Ks[(t * 16 + lane15) * 64];
            Frag16 kf0, kf1;
            kf0.u[0] = *(const uint4*)&krow[h8 * 8];
            kf0.u[1] = *(const uint4*)&krow[16 + h8 * 8];
            kf1.u[0] = *(const uint4*)&krow[32 + h8 * 8];
            kf1.u[1] = *(const uint4*)&krow[48 + h8 * 8];
            sacc[t] = __builtin_amdgcn_wmma_f32_16x16x32_bf16(
                false, qf0.v, false, kf0.v, (short)0, sacc[t], false, false);
            sacc[t] = __builtin_amdgcn_wmma_f32_16x16x32_bf16(
                false, qf1.v, false, kf1.v, (short)0, sacc[t], false, false);
        }

        // online softmax: row max (cross-lane within 16-lane half = one C-row)
        float alpha[8];
#pragma unroll
        for (int r = 0; r < 8; ++r) {
            float mx = sacc[0][r];
#pragma unroll
            for (int t = 1; t < 8; ++t) mx = fmaxf(mx, sacc[t][r]);
            mx = fmaxf(mx, __shfl_xor(mx, 1, 32));
            mx = fmaxf(mx, __shfl_xor(mx, 2, 32));
            mx = fmaxf(mx, __shfl_xor(mx, 4, 32));
            mx = fmaxf(mx, __shfl_xor(mx, 8, 32));
            float mn = fmaxf(m_i[r], mx);
            alpha[r] = __expf(m_i[r] - mn);
            m_i[r]   = mn;
        }

        // P = exp(S - m), row sums, stash P (bf16) in per-wave LDS slab
        __bf16* pslab = &Ps[wid * 16 * 128];
        float rs[8];
#pragma unroll
        for (int r = 0; r < 8; ++r) rs[r] = 0.0f;
#pragma unroll
        for (int t = 0; t < 8; ++t)
#pragma unroll
            for (int r = 0; r < 8; ++r) {
                float p = __expf(sacc[t][r] - m_i[r]);
                rs[r] += p;
                pslab[(r + 8 * h8) * 128 + t * 16 + lane15] = (__bf16)p;
            }
#pragma unroll
        for (int r = 0; r < 8; ++r) {
            float s = rs[r];
            s += __shfl_xor(s, 1, 32);
            s += __shfl_xor(s, 2, 32);
            s += __shfl_xor(s, 4, 32);
            s += __shfl_xor(s, 8, 32);
            l_i[r] = l_i[r] * alpha[r] + s;
        }
#pragma unroll
        for (int dt = 0; dt < 4; ++dt)
#pragma unroll
            for (int r = 0; r < 8; ++r) o[dt][r] *= alpha[r];

        // O += P @ V  (K dim = 128 -> 4 chunks of 32)
        Frag16 pA[4];
        const __bf16* prow = &Ps[wid * 16 * 128 + lane15 * 128];
#pragma unroll
        for (int kc = 0; kc < 4; ++kc) {
            pA[kc].u[0] = *(const uint4*)&prow[kc * 32 + h8 * 8];
            pA[kc].u[1] = *(const uint4*)&prow[kc * 32 + 16 + h8 * 8];
        }
#pragma unroll
        for (int dt = 0; dt < 4; ++dt) {
            const __bf16* vrow = &Vs[(dt * 16 + lane15) * 128];
#pragma unroll
            for (int kc = 0; kc < 4; ++kc) {
                Frag16 vb;
                vb.u[0] = *(const uint4*)&vrow[kc * 32 + h8 * 8];
                vb.u[1] = *(const uint4*)&vrow[kc * 32 + 16 + h8 * 8];
                o[dt] = __builtin_amdgcn_wmma_f32_16x16x32_bf16(
                    false, pA[kc].v, false, vb.v, (short)0, o[dt], false, false);
            }
        }
    }

    // finalize: O / l, write bf16 into [B, N, C] for the projection GEMM
#pragma unroll
    for (int r = 0; r < 8; ++r) l_i[r] = 1.0f / l_i[r];
#pragma unroll
    for (int dt = 0; dt < 4; ++dt)
#pragma unroll
        for (int r = 0; r < 8; ++r) {
            int srow = qr0 + r + 8 * h8;
            obuf[((size_t)b * SEQ + srow) * DIMC + h * HDIM + dt * 16 + lane15] =
                (__bf16)(o[dt][r] * l_i[r]);
        }
}

// ---------------------------------------------------------------- launch
extern "C" void kernel_launch(void* const* d_in, const int* in_sizes, int n_in,
                              void* d_out, int out_size, void* d_ws, size_t ws_size,
                              hipStream_t stream) {
    const float* x      = (const float*)d_in[0];
    const float* qkv_w  = (const float*)d_in[1];
    const float* qkv_b  = (const float*)d_in[2];
    const float* proj_w = (const float*)d_in[3];
    const float* proj_b = (const float*)d_in[4];
    float*       out    = (float*)d_out;
    char*        ws     = (char*)d_ws;

    __bf16* xb    = (__bf16*)(ws + XB_OFF);
    __bf16* wqkv  = (__bf16*)(ws + WQKV_OFF);
    __bf16* wproj = (__bf16*)(ws + WPRJ_OFF);
    __bf16* qkv   = (__bf16*)(ws + QKV_OFF);   // q | k | vT
    __bf16* obuf  = (__bf16*)(ws + OBUF_OFF);

    // 1) fp32 -> bf16 conversions
    {
        int n;
        n = MROWS * DIMC;
        f32_to_bf16_kernel<<<(n / 4 + 255) / 256, 256, 0, stream>>>(x, xb, n);
        n = K3 * DIMC;
        f32_to_bf16_kernel<<<(n / 4 + 255) / 256, 256, 0, stream>>>(qkv_w, wqkv, n);
        n = DIMC * DIMC;
        f32_to_bf16_kernel<<<(n / 4 + 255) / 256, 256, 0, stream>>>(proj_w, wproj, n);
    }

    // 2) QKV GEMM + scatter epilogue (q scaled, v transposed)
    gemm_bf16_wmma<1><<<dim3(K3 / 128, MROWS / 128), 256, 0, stream>>>(
        xb, wqkv, qkv_b, nullptr, qkv, MROWS, K3, DIMC);

    // 3) fused flash attention
    flash_attn_wmma<<<dim3(SEQ / 128, BATCH * NHEADS), 256, 0, stream>>>(
        qkv, qkv + QKV_ELEMS, qkv + 2 * QKV_ELEMS, obuf);

    // 4) output projection GEMM -> fp32 d_out
    gemm_bf16_wmma<0><<<dim3(DIMC / 128, MROWS / 128), 256, 0, stream>>>(
        obuf, wproj, proj_b, out, nullptr, MROWS, DIMC, DIMC);
}
